// OrientationNet_86131274154204
// MI455X (gfx1250) — matrix operand order repaired and checked by reading, not compiled, not run on hardware
//
#include <hip/hip_runtime.h>

typedef __attribute__((ext_vector_type(16))) _Float16 v16h;
typedef __attribute__((ext_vector_type(8)))  float    v8f;

#define NIMG   16
#define OWID   62
#define NPIX   3844        // 62*62
#define NTILES 241         // ceil(3844/16)
#define L2E    1.44269504f

__device__ __forceinline__ float log2sig(float z) {
    // log2(sigmoid(z)), z in natural-log domain
    float e = __builtin_amdgcn_exp2f(-z * L2E);
    return -__builtin_amdgcn_logf(1.0f + e);
}

// Build the binary-channel GEMM operands:
//   Bbuf (f16, WMMA 32x16 B layout, 8 column tiles): Delta[k][c] = log2sig(10(W-q)) - log2sig(-10q)
//   bias[c] = sum_k log2sig(-10q)          (c = d*4+j, 128 columns)
// Also zero-initializes d_out (atomic accumulation target).
__global__ void orient_prep(const float* __restrict__ W, const float* __restrict__ q,
                            _Float16* __restrict__ Bbuf, float* __restrict__ bias,
                            float* __restrict__ out) {
    int tid = threadIdx.x;
    for (int i = tid; i < 8 * 32 * 16; i += 256) Bbuf[i] = (_Float16)0.0f;
    if (tid < 64) out[tid] = 0.0f;
    __syncthreads();
    if (tid < 128) {
        int d = tid >> 2, j = tid & 3;
        int c = d * 4 + j;
        int tile = c >> 4, cl = c & 15;
        float bsum = 0.0f;
        for (int k = 0; k < 9; ++k) {
            float Wv = W[(d * 18 + k) * 4 + j] * 10.0f;
            float qv = q[(d * 18 + k) * 4 + j] * 10.0f;
            float t0 = log2sig(-qv);
            float t1 = log2sig(Wv - qv);
            bsum += t0;
            // B layout (32x16 f16): lanes 0-15 = col, halves e = K rows 0-15; K>=9 stays zero
            Bbuf[(tile * 32 + cl) * 16 + k] = (_Float16)(t1 - t0);
        }
        bias[c] = bsum;
    }
}

__global__ __launch_bounds__(256) void orient_main(
        const float* __restrict__ x, const float* __restrict__ W,
        const float* __restrict__ q, const _Float16* __restrict__ Bbuf,
        const float* __restrict__ bias, float* __restrict__ out) {
    __shared__ __align__(32) _Float16 sB[8 * 32 * 16]; // 8 KB  : WMMA B tiles
    __shared__ float sCoef[32 * 4 * 24];               // 12 KB : per (d,j): 10*L2E*{W,q}[m=0..8]
    __shared__ float sBias[128];
    __shared__ float sP[8][16][12];                    // 6 KB  : patch values per wave
    __shared__ float sRed[4];

    const int tid  = threadIdx.x;
    const int b    = blockIdx.x;
    const int wave = tid >> 5;
    const int lane = tid & 31;
    const int l15  = lane & 15;

    // ---- stage constants into LDS ----
    {
        const unsigned int* src = (const unsigned int*)Bbuf;
        unsigned int* dst = (unsigned int*)sB;
        for (int i = tid; i < 2048; i += 256) dst[i] = src[i];
    }
    for (int i = tid; i < 2304; i += 256) {            // 128 (d,j) * 18 coefficients
        int dj = i / 18, r = i % 18;
        int d = dj >> 2, j = dj & 3;
        int m = (r < 9) ? r : (r - 9);
        const float* s = (r < 9) ? W : q;
        float v = s[(d * 18 + 9 + m) * 4 + j] * (10.0f * L2E);
        sCoef[dj * 24 + ((r < 9) ? r : (r + 3))] = v;  // W at +0..8, q at +12..20
    }
    if (tid < 128) sBias[tid] = bias[tid];
    if (tid < 4) sRed[tid] = 0.0f;

    // ---- per-pixel 3x3 patch (lane m and m+16 handle the same pixel) ----
    int tile_t = blockIdx.y * 8 + wave;
    const bool wact = (tile_t < NTILES);
    const int tt = wact ? tile_t : 0;
    const int pixBase = tt * 16;
    int myPix = pixBase + l15;
    int pi = (myPix < NPIX) ? myPix : (NPIX - 1);
    int oy = pi / OWID;
    int ox = pi - oy * OWID;
    const float* xb = x + (b << 12) + oy * 64 + ox;

    float p[9];
    p[0] = xb[0];   p[1] = xb[1];   p[2] = xb[2];
    p[3] = xb[64];  p[4] = xb[65];  p[5] = xb[66];
    p[6] = xb[128]; p[7] = xb[129]; p[8] = xb[130];
    float ctr = p[4];
    float x1[9];
    #pragma unroll
    for (int m = 0; m < 9; ++m) x1[m] = (fabsf(p[m] - ctr) <= 3.0f) ? 1.0f : 0.0f;
    // x1[4] == 1 by construction

    if (lane < 16) {
        #pragma unroll
        for (int m = 0; m < 9; ++m) sP[wave][l15][m] = p[m];
    }
    __syncthreads();

    // ---- binary channels: D = X1(16x32 f16) x Delta(32x16 f16) + bias, 8 column tiles ----
    // A layout: lanes 0-15 hold K=0..8 (halves 0..8), lanes 16-31 hold zeros
    const bool lo = (lane < 16);
    v16h A = {};
    A[0] = (_Float16)(lo ? x1[0] : 0.0f);
    #pragma unroll
    for (int e = 1; e < 9; ++e) A[e] = (_Float16)(lo ? x1[e] : 0.0f);

    v8f Dt[8];
    const v16h* sBv = (const v16h*)sB;
    #pragma unroll
    for (int t = 0; t < 8; ++t) {
        v16h Bm = sBv[t * 32 + lane];
        float cb = sBias[t * 16 + l15];
        v8f C = {cb, cb, cb, cb, cb, cb, cb, cb};
        Dt[t] = __builtin_amdgcn_wmma_f32_16x16x32_f16(false, A, false, Bm,
                                                       (short)0, C, false, false);
    }

    // ---- continuous channels + Sum_d: lane l owns j = l&3, d in {t*4 + (l15>>2)} ----
    const int dq   = l15 >> 2;
    const int j    = l15 & 3;
    const int half = lane >> 4;       // D rows: 0 -> pixels 0..7, 1 -> pixels 8..15
    float menb[8];
    #pragma unroll
    for (int r = 0; r < 8; ++r) menb[r] = 0.0f;

    #pragma unroll
    for (int t = 0; t < 8; ++t) {
        int d = t * 4 + dq;
        const float* cf = &sCoef[(d * 4 + j) * 24];
        float wv[9], qv[9];
        #pragma unroll
        for (int m = 0; m < 9; ++m) { wv[m] = cf[m]; qv[m] = cf[12 + m]; }
        #pragma unroll
        for (int r = 0; r < 8; ++r) {
            const float* pp = &sP[wave][half * 8 + r][0];
            float prod = __builtin_amdgcn_exp2f(Dt[t][r]);   // exp2 of binary log-sum (+bias)
            #pragma unroll
            for (int m = 0; m < 9; ++m) {
                float z = qv[m] - pp[m] * wv[m];             // = -(10*(p*W - q))*log2(e)
                prod *= __builtin_amdgcn_rcpf(1.0f + __builtin_amdgcn_exp2f(z));
            }
            menb[r] += prod;
        }
    }
    // complete Sum over all 32 d: combine the 4 lanes sharing j (l15 in {j, j+4, j+8, j+12})
    #pragma unroll
    for (int r = 0; r < 8; ++r) {
        menb[r] += __shfl_xor(menb[r], 4, 32);
        menb[r] += __shfl_xor(menb[r], 8, 32);
    }

    float acc = 0.0f;
    if (dq == 0) {   // lanes 0-3 (pixels 0-7) and 16-19 (pixels 8-15), j = lane&3
        #pragma unroll
        for (int r = 0; r < 8; ++r) {
            int pix = pixBase + half * 8 + r;
            float valid = (wact && pix < NPIX) ? 1.0f : 0.0f;
            float soma = __builtin_amdgcn_rcpf(
                1.0f + __builtin_amdgcn_exp2f((0.5f - menb[r]) * (10.0f * L2E)));
            acc += soma * valid;
        }
        atomicAdd(&sRed[j], acc);
    }
    __syncthreads();
    if (tid < 4) atomicAdd(&out[b * 4 + tid], sRed[tid]);
}

extern "C" void kernel_launch(void* const* d_in, const int* in_sizes, int n_in,
                              void* d_out, int out_size, void* d_ws, size_t ws_size,
                              hipStream_t stream) {
    const float* x = (const float*)d_in[0];   // (16,1,64,64)
    const float* W = (const float*)d_in[1];   // (32,18,4)
    const float* q = (const float*)d_in[2];   // (32,18,4)
    float* out = (float*)d_out;               // (16,4)
    _Float16* Bbuf = (_Float16*)d_ws;                      // 8 KB f16 B tiles
    float* bias = (float*)((char*)d_ws + 8192);            // 512 B bias

    orient_prep<<<1, 256, 0, stream>>>(W, q, Bbuf, bias, out);
    dim3 grid(NIMG, 31);                                   // 31*8 waves >= 241 tiles/image
    orient_main<<<grid, 256, 0, stream>>>(x, W, q, Bbuf, bias, out);
}
